// SelfAttention_28827820490965
// MI455X (gfx1250) — matrix operand order repaired and checked by reading, not compiled, not run on hardware
//
#include <hip/hip_runtime.h>
#include <hip/hip_bf16.h>

#define DIM  256
#define NSEQ 8192

typedef __attribute__((ext_vector_type(16))) __bf16       v16bf;
typedef __attribute__((ext_vector_type(2)))  __bf16       v2bf;
typedef __attribute__((ext_vector_type(8)))  float        v8f;
typedef __attribute__((ext_vector_type(4)))  float        v4f;
typedef __attribute__((ext_vector_type(4)))  unsigned int v4u;

union ABfrag {
    v16bf v;
    v4u   q[2];
    unsigned short h[16];
};

__device__ inline unsigned short f2bf(float f) {
    unsigned u = __float_as_uint(f);
    u += 0x7FFFu + ((u >> 16) & 1u);   // round-to-nearest-even
    return (unsigned short)(u >> 16);
}

__device__ inline unsigned cvt2(float a, float b) {
#if __has_builtin(__builtin_amdgcn_cvt_pk_bf16_f32)
    union { v2bf v; unsigned u; } cv;
    cv.v = __builtin_amdgcn_cvt_pk_bf16_f32(a, b);
    return cv.u;
#else
    return (unsigned)f2bf(a) | ((unsigned)f2bf(b) << 16);
#endif
}

__device__ inline v8f wmma_bf16(v16bf a, v16bf b, v8f c) {
    return __builtin_amdgcn_wmma_f32_16x16x32_bf16(
        false, a, false, b, (short)0, c, false, false);
}

// DPP16 cross-lane reduction within each 16-lane row (never crosses rows).
template <int CTRL>
__device__ inline float dpp_movf(float x) {
    return __int_as_float(__builtin_amdgcn_update_dpp(
        0, __float_as_int(x), CTRL, 0xF, 0xF, true));
}
__device__ inline float row_max16(float x) {
    x = fmaxf(x, dpp_movf<0xB1>(x));    // quad_perm(1,0,3,2)  == xor 1
    x = fmaxf(x, dpp_movf<0x4E>(x));    // quad_perm(2,3,0,1)  == xor 2
    x = fmaxf(x, dpp_movf<0x141>(x));   // row_half_mirror     == xor 7
    x = fmaxf(x, dpp_movf<0x140>(x));   // row_mirror          == xor 15
    return x;
}
__device__ inline float row_sum16(float x) {
    x += dpp_movf<0xB1>(x);
    x += dpp_movf<0x4E>(x);
    x += dpp_movf<0x141>(x);
    x += dpp_movf<0x140>(x);
    return x;
}

// CDNA5 async global->LDS copy (ASYNCcnt) -- 16B per lane.
__device__ inline void async_load_b128(unsigned short* lds_dst, const void* gsrc) {
    const unsigned loff = (unsigned)(uintptr_t)lds_dst;    // low 32b = LDS offset
    asm volatile("global_load_async_to_lds_b128 %0, %1, off"
                 :: "v"(loff), "v"(gsrc) : "memory");
}
__device__ inline void wait_async0() {
    asm volatile("s_wait_asynccnt 0x0" ::: "memory");
}

// ---------------------------------------------------------------------------
// Kernel 0: one-shot fp32 -> bf16 conversion (8 elements / thread).
// ---------------------------------------------------------------------------
__global__ __launch_bounds__(256) void cvt_bf16_kernel(
    const float* __restrict__ src, unsigned short* __restrict__ dst, int n)
{
    const int idx = (blockIdx.x * 256 + threadIdx.x) * 8;
    if (idx >= n) return;
    v4f a = *(const v4f*)(src + idx);
    v4f b = *(const v4f*)(src + idx + 4);
    v4u o;
    o[0] = cvt2(a[0], a[1]);
    o[1] = cvt2(a[2], a[3]);
    o[2] = cvt2(b[0], b[1]);
    o[3] = cvt2(b[2], b[3]);
    *(v4u*)(dst + idx) = o;
}

// ---------------------------------------------------------------------------
// Kernel 1: QKV projection, pure bf16 WMMA.  y = X @ W.T
// ---------------------------------------------------------------------------
__global__ __launch_bounds__(256) void qkv_proj_kernel(
    const unsigned short* __restrict__ Xb,  const unsigned short* __restrict__ Wqb,
    const unsigned short* __restrict__ Wkb, const unsigned short* __restrict__ Wvb,
    unsigned short* __restrict__ Qb, unsigned short* __restrict__ Kb,
    unsigned short* __restrict__ Vt)
{
    const int lane = threadIdx.x & 31;
    const int wave = blockIdx.x * (blockDim.x >> 5) + (threadIdx.x >> 5);

    const int tiles_per_mat = (NSEQ / 16) * (DIM / 16);   // 8192
    const int which = wave / tiles_per_mat;               // 0=Q 1=K 2=V
    const int rem   = wave % tiles_per_mat;
    const int i     = rem / (DIM / 16);
    const int j     = rem % (DIM / 16);

    const unsigned short* W = (which == 0) ? Wqb : (which == 1) ? Wkb : Wvb;

    const int nl   = lane & 15;
    const int aoff = (lane < 16) ? 0 : 16;
    const int boff = (lane < 16) ? 0 : 16;
    const unsigned short* xrow = Xb + (size_t)(i * 16 + nl) * DIM;
    const unsigned short* wrow = W  + (size_t)(j * 16 + nl) * DIM + boff;

    v8f acc = {};
#pragma unroll
    for (int kc = 0; kc < DIM / 32; ++kc) {
        ABfrag A, B;
        A.q[0] = *(const v4u*)(xrow + kc * 32 + aoff);
        A.q[1] = *(const v4u*)(xrow + kc * 32 + aoff + 8);
        B.q[0] = *(const v4u*)(wrow + kc * 32);
        B.q[1] = *(const v4u*)(wrow + kc * 32 + 8);
        acc = wmma_bf16(A.v, B.v, acc);
    }

    if (which == 0) {
#pragma unroll
        for (int r = 0; r < 8; ++r) acc[r] *= 0.0625f;   // 1/sqrt(256)
    }

    if (which < 2) {
        unsigned short* outp = (which == 0) ? Qb : Kb;
        const int col   = j * 16 + nl;
        const int rbase = i * 16 + ((lane < 16) ? 0 : 8);
#pragma unroll
        for (int r = 0; r < 8; ++r)
            outp[(size_t)(rbase + r) * DIM + col] = f2bf(acc[r]);
    } else {
        const int d  = j * 16 + nl;
        const int nb = i * 16 + ((lane < 16) ? 0 : 8);
        union { v4u q; unsigned u[4]; } pack;
#pragma unroll
        for (int e = 0; e < 4; ++e) pack.u[e] = cvt2(acc[2 * e], acc[2 * e + 1]);
        *(v4u*)(Vt + (size_t)d * NSEQ + nb) = pack.q;
    }
}

// ---------------------------------------------------------------------------
// Kernel 2: fused flash attention with async double-buffered K/V staging.
// Workgroup = 8 waves; K/V tiles for each 64-key chunk are loaded once into
// LDS via global_load_async_to_lds_b128 (copy of chunk i+1 overlaps compute
// of chunk i).  Each wave owns 16 query rows.
// Dynamic LDS: K 2x[64][256] (64KB) | V 2x[256][64] (64KB) | P 8x[16][64] (16KB)
// ---------------------------------------------------------------------------
__global__ __launch_bounds__(256) void flash_attn_kernel(
    const unsigned short* __restrict__ Qb, const unsigned short* __restrict__ Kb,
    const unsigned short* __restrict__ Vt, float* __restrict__ Out)
{
    extern __shared__ unsigned short smem[];
    unsigned short* kls = smem;                             // 2 * 64*256
    unsigned short* vls = smem + 2 * 64 * DIM;              // 2 * 256*64
    unsigned short* pls = smem + 2 * 64 * DIM + 2 * 256 * 64;

    const int lane  = threadIdx.x & 31;
    const int wl    = threadIdx.x >> 5;
    const int qrow0 = (blockIdx.x * 8 + wl) * 16;

    const int nl   = lane & 15;
    const int aoff = (lane < 16) ? 0 : 16;
    const int boff = (lane < 16) ? 0 : 16;

    // --- async staging helpers (each wave covers 1/8 of the tile) ---
    const int krow  = wl * 8;                 // K rows this wave stages
    const int kcol  = lane * 8;               // 16B per lane, 512B row
    const int vrow0 = wl * 32 + (lane >> 3) ; // V: 4 rows per issue
    const int vcol  = (lane & 7) * 8;         // 16B per lane, 128B row

    // stage chunk 0 into buffer 0
#pragma unroll
    for (int r = 0; r < 8; ++r)
        async_load_b128(kls + (krow + r) * DIM + kcol,
                        Kb + (size_t)(krow + r) * DIM + kcol);
#pragma unroll
    for (int r = 0; r < 8; ++r)
        async_load_b128(vls + (vrow0 + 4 * r) * 64 + vcol,
                        Vt + (size_t)(vrow0 + 4 * r) * NSEQ + vcol);
    wait_async0();
    __syncthreads();

    // Q A-fragments resident for the whole pass
    ABfrag qa[8];
    const unsigned short* qrow = Qb + (size_t)(qrow0 + nl) * DIM;
#pragma unroll
    for (int kc = 0; kc < 8; ++kc) {
        qa[kc].q[0] = *(const v4u*)(qrow + kc * 32 + aoff);
        qa[kc].q[1] = *(const v4u*)(qrow + kc * 32 + aoff + 8);
    }

    float mr[8], lr[8];
#pragma unroll
    for (int r = 0; r < 8; ++r) { mr[r] = -1e30f; lr[r] = 0.f; }
    v8f O[16];
#pragma unroll
    for (int t = 0; t < 16; ++t) O[t] = (v8f){};

    unsigned short* myl = pls + wl * (16 * 64);
    const int rowb = (lane < 16) ? 0 : 8;

    for (int j0 = 0; j0 < NSEQ; j0 += 64) {
        const int cur = (j0 >> 6) & 1;
        unsigned short* kcur = kls + cur * (64 * DIM);
        unsigned short* vcur = vls + cur * (256 * 64);

        // issue async copy of next chunk into the other buffer
        if (j0 + 64 < NSEQ) {
            unsigned short* knxt = kls + (cur ^ 1) * (64 * DIM);
            unsigned short* vnxt = vls + (cur ^ 1) * (256 * 64);
#pragma unroll
            for (int r = 0; r < 8; ++r)
                async_load_b128(knxt + (krow + r) * DIM + kcol,
                                Kb + (size_t)(j0 + 64 + krow + r) * DIM + kcol);
#pragma unroll
            for (int r = 0; r < 8; ++r)
                async_load_b128(vnxt + (vrow0 + 4 * r) * 64 + vcol,
                                Vt + (size_t)(vrow0 + 4 * r) * NSEQ + j0 + 64 + vcol);
        }

        // ---- S = Q @ K.T for 64 keys (four 16x16 tiles), K from LDS ----
        v8f s0 = {}, s1 = {}, s2 = {}, s3 = {};
        const unsigned short* kp = kcur + (size_t)nl * DIM + boff;
#pragma unroll
        for (int kc = 0; kc < 8; ++kc) {
            ABfrag b0, b1, b2, b3;
            b0.q[0] = *(const v4u*)(kp + kc * 32);
            b0.q[1] = *(const v4u*)(kp + kc * 32 + 8);
            b1.q[0] = *(const v4u*)(kp + 16 * DIM + kc * 32);
            b1.q[1] = *(const v4u*)(kp + 16 * DIM + kc * 32 + 8);
            b2.q[0] = *(const v4u*)(kp + 32 * DIM + kc * 32);
            b2.q[1] = *(const v4u*)(kp + 32 * DIM + kc * 32 + 8);
            b3.q[0] = *(const v4u*)(kp + 48 * DIM + kc * 32);
            b3.q[1] = *(const v4u*)(kp + 48 * DIM + kc * 32 + 8);
            s0 = wmma_bf16(qa[kc].v, b0.v, s0);
            s1 = wmma_bf16(qa[kc].v, b1.v, s1);
            s2 = wmma_bf16(qa[kc].v, b2.v, s2);
            s3 = wmma_bf16(qa[kc].v, b3.v, s3);
        }

        // ---- online softmax ----
        float c[8];
#pragma unroll
        for (int r = 0; r < 8; ++r) {
            float tm = fmaxf(fmaxf(s0[r], s1[r]), fmaxf(s2[r], s3[r]));
            tm = row_max16(tm);
            float mn = fmaxf(mr[r], tm);
            c[r] = __expf(mr[r] - mn);
            float p0 = __expf(s0[r] - mn);
            float p1 = __expf(s1[r] - mn);
            float p2 = __expf(s2[r] - mn);
            float p3 = __expf(s3[r] - mn);
            mr[r] = mn;
            s0[r] = p0; s1[r] = p1; s2[r] = p2; s3[r] = p3;
            float rs = row_sum16((p0 + p1) + (p2 + p3));
            lr[r] = lr[r] * c[r] + rs;
        }
#pragma unroll
        for (int t = 0; t < 16; ++t)
#pragma unroll
            for (int r = 0; r < 8; ++r) O[t][r] *= c[r];

        // ---- stage P (C-layout -> row-major LDS -> A-layout) ----
#pragma unroll
        for (int r = 0; r < 8; ++r) {
            unsigned short* lp = myl + (rowb + r) * 64 + nl;
            lp[0]  = f2bf(s0[r]);
            lp[16] = f2bf(s1[r]);
            lp[32] = f2bf(s2[r]);
            lp[48] = f2bf(s3[r]);
        }
        ABfrag pf0, pf1;
        pf0.q[0] = *(const v4u*)(myl + nl * 64 + aoff);
        pf0.q[1] = *(const v4u*)(myl + nl * 64 + aoff + 8);
        pf1.q[0] = *(const v4u*)(myl + nl * 64 + 32 + aoff);
        pf1.q[1] = *(const v4u*)(myl + nl * 64 + 32 + aoff + 8);

        // ---- O += P @ V, V from LDS ----
#pragma unroll
        for (int t = 0; t < 16; ++t) {
            const unsigned short* vp = vcur + (size_t)(t * 16 + nl) * 64 + boff;
            ABfrag vf0, vf1;
            vf0.q[0] = *(const v4u*)(vp);
            vf0.q[1] = *(const v4u*)(vp + 8);
            vf1.q[0] = *(const v4u*)(vp + 32);
            vf1.q[1] = *(const v4u*)(vp + 40);
            O[t] = wmma_bf16(pf0.v, vf0.v, O[t]);
            O[t] = wmma_bf16(pf1.v, vf1.v, O[t]);
        }

        // own async copies done + whole workgroup done loading/reading
        wait_async0();
        __syncthreads();
    }

    // ---- normalize (1 rcp per row) and store fp32 output ----
    const int orow = qrow0 + rowb;
#pragma unroll
    for (int r = 0; r < 8; ++r) {
        const float inv = 1.0f / lr[r];
#pragma unroll
        for (int t = 0; t < 16; ++t)
            Out[(size_t)(orow + r) * DIM + t * 16 + nl] = O[t][r] * inv;
    }
}

extern "C" void kernel_launch(void* const* d_in, const int* in_sizes, int n_in,
                              void* d_out, int out_size, void* d_ws, size_t ws_size,
                              hipStream_t stream) {
    const float* X  = (const float*)d_in[0];
    const float* Wq = (const float*)d_in[1];
    const float* Wk = (const float*)d_in[2];
    const float* Wv = (const float*)d_in[3];
    float* Out = (float*)d_out;

    // workspace layout (bf16): Qb | Kb | Vt | Xb | Wqb | Wkb | Wvb
    unsigned short* Qb  = (unsigned short*)d_ws;
    unsigned short* Kb  = Qb  + (size_t)NSEQ * DIM;
    unsigned short* Vt  = Kb  + (size_t)NSEQ * DIM;
    unsigned short* Xb  = Vt  + (size_t)NSEQ * DIM;
    unsigned short* Wqb = Xb  + (size_t)NSEQ * DIM;
    unsigned short* Wkb = Wqb + (size_t)DIM * DIM;
    unsigned short* Wvb = Wkb + (size_t)DIM * DIM;

    const int nX = NSEQ * DIM;
    const int nW = DIM * DIM;
    cvt_bf16_kernel<<<nX / 2048, 256, 0, stream>>>(X,  Xb,  nX);
    cvt_bf16_kernel<<<nW / 2048, 256, 0, stream>>>(Wq, Wqb, nW);
    cvt_bf16_kernel<<<nW / 2048, 256, 0, stream>>>(Wk, Wkb, nW);
    cvt_bf16_kernel<<<nW / 2048, 256, 0, stream>>>(Wv, Wvb, nW);

    const int tiles = 3 * (NSEQ / 16) * (DIM / 16);
    qkv_proj_kernel<<<tiles / 8, 256, 0, stream>>>(Xb, Wqb, Wkb, Wvb, Qb, Kb, Vt);

    // dynamic LDS: 2*64*256 + 2*256*64 + 8*16*64 bf16 elements = 144 KB
    const size_t lds_bytes = (2 * 64 * DIM + 2 * 256 * 64 + 8 * 16 * 64) * sizeof(unsigned short);
    flash_attn_kernel<<<NSEQ / 16 / 8, 256, lds_bytes, stream>>>(Qb, Kb, Vt, Out);
}
